// MAB_936302870610
// MI455X (gfx1250) — matrix operand order repaired and checked by reading, not compiled, not run on hardware
//
#include <hip/hip_runtime.h>
#include <hip/hip_bf16.h>

// MAB forward for MI455X (gfx1250, wave32, WMMA 16x16x32 bf16).
// [transpose W -> bf16 Wt] -> [proj q/k/v] -> [softmax P] -> [transpose V]
// -> [P@V] -> [Wo proj] -> [LayerNorm]
// GEMMs: 128x256 macro-tile, 64x64 wave tiles, double-buffered LDS staging.
// Staging: GLOBAL_LOAD_ASYNC_TO_LDS_B128 (ASYNCcnt); pv_gemm additionally uses
// the Tensor Data Mover (TENSOR_LOAD_TO_LDS + s_wait_tensorcnt) when available.

typedef __attribute__((ext_vector_type(16))) __bf16 v16bf;
typedef __attribute__((ext_vector_type(8)))  __bf16 v8bf;
typedef __attribute__((ext_vector_type(8)))  float  v8f;
typedef __attribute__((ext_vector_type(4)))  int    v4i;
typedef __attribute__((ext_vector_type(4)))  unsigned u32x4;
typedef __attribute__((ext_vector_type(8)))  int    i32x8;
typedef __attribute__((ext_vector_type(4)))  int    i32x4;

#define NB 8
#define NQ 2048
#define NKK 2048
#define DVV 1024

#if defined(__has_builtin)
#if __has_builtin(__builtin_amdgcn_global_load_async_to_lds_b128) && \
    __has_builtin(__builtin_amdgcn_s_wait_asynccnt)
#define HAVE_ASYNC_COPY 1
#endif
#if __has_builtin(__builtin_amdgcn_s_wait_tensorcnt)
#if __has_builtin(__builtin_amdgcn_tensor_load_to_lds_d2)
#define HAVE_TDM 1
#define TDM_LOAD(g0, g1) __builtin_amdgcn_tensor_load_to_lds_d2((g0), (g1), 0)
#elif __has_builtin(__builtin_amdgcn_tensor_load_to_lds)
// This toolchain's form takes 6 args: (g0, g1, g2, g3, g4, cpol).
#define HAVE_TDM 1
#define TDM_LOAD(g0, g1)                                              \
  __builtin_amdgcn_tensor_load_to_lds((g0), (g1), (i32x4)(0),         \
                                      (i32x4)(0), (i32x8)(0), 0)
#endif
#endif
#endif

#ifdef HAVE_ASYNC_COPY
__device__ __forceinline__ void lds_copy16(void* l, const void* g) {
  __builtin_amdgcn_global_load_async_to_lds_b128((v4i*)g, (v4i*)l, 0, 0);
}
__device__ __forceinline__ void async_join() {
  __builtin_amdgcn_s_wait_asynccnt(0);
}
#else
__device__ __forceinline__ void lds_copy16(void* l, const void* g) {
  *(v8bf*)l = *(const v8bf*)g;
}
__device__ __forceinline__ void async_join() {}
#endif

#ifdef HAVE_TDM
// D# group 0 (ISA 8.3): [1:0]=count=1, [63:32]=lds_addr, [120:64]=global_addr,
// [127:126]=type=2.
__device__ __forceinline__ u32x4 tdm_g0(unsigned lds_off,
                                        unsigned long long gaddr) {
  u32x4 g;
  g[0] = 1u;
  g[1] = lds_off;
  g[2] = (unsigned)gaddr;
  g[3] = ((unsigned)(gaddr >> 32) & 0x1FFFFFFu) | (2u << 30);
  return g;
}
// D# group 1 (ISA 8.4): 2D tile (tile0 x tile1 bf16 elems), row stride s0 elems;
// pad: insert 4 DWORDs (8 bf16) after every 16 DWORDs (32 bf16) -> LDS row 40.
__device__ __forceinline__ i32x8 tdm_g1(unsigned tdim0, unsigned tdim1,
                                        unsigned tile0, unsigned tile1,
                                        unsigned long long s0, bool pad) {
  i32x8 g = {};
  unsigned w0 = (1u << 16);                              // data_size = 2 bytes
  if (pad) w0 |= (1u << 20) | (3u << 22) | (3u << 25);   // pad_en, 16DW, 4DW
  g[0] = (int)w0;
  g[1] = (int)((tdim0 & 0xFFFFu) << 16);                 // tensor_dim0[15:0]
  g[2] = (int)(((tdim0 >> 16) & 0xFFFFu) | ((tdim1 & 0xFFFFu) << 16));
  g[3] = (int)(((tdim1 >> 16) & 0xFFFFu) | ((tile0 & 0xFFFFu) << 16));
  g[4] = (int)(tile1 & 0xFFFFu);                         // tile_dim1 (tile_dim2=0)
  g[5] = (int)(unsigned)s0;                              // dim0 stride lo32
  g[6] = (int)((unsigned)(s0 >> 32) & 0xFFFFu);          // dim0 stride hi16
  g[7] = 0;
  return g;
}
#endif

__device__ __forceinline__ __bf16 f32_to_bf16(float f) {
  union { float f; unsigned u; } v; v.f = f;
  unsigned r = v.u + 0x7FFFu + ((v.u >> 16) & 1u);   // round-to-nearest-even
  unsigned short h = (unsigned short)(r >> 16);
  return __builtin_bit_cast(__bf16, h);
}

__device__ __forceinline__ v16bf concat8(v8bf lo, v8bf hi) {
  return __builtin_shufflevector(lo, hi, 0, 1, 2, 3, 4, 5, 6, 7,
                                 8, 9, 10, 11, 12, 13, 14, 15);
}

// A fragment: 16(M)x32(K), element (m,k) at A[m*lda + k]; lda % 8 == 0, 16B-aligned.
__device__ __forceinline__ v16bf load_a_frag(const __bf16* A, int lda) {
  int lane = threadIdx.x & 31;
  int m = lane & 15, hi = lane >> 4;
  const __bf16* p = A + m * lda + 16 * hi;
  return concat8(*(const v8bf*)p, *(const v8bf*)(p + 8));
}

// B fragment from k-contiguous (transposed) source: B(k,n) = T[n*ldt + k].
__device__ __forceinline__ v16bf load_bT_frag(const __bf16* T, int ldt) {
  int lane = threadIdx.x & 31;
  int n = lane & 15, hi = lane >> 4;
  const __bf16* p = T + n * ldt + 16 * hi;
  return concat8(*(const v8bf*)p, *(const v8bf*)(p + 8));
}

// B fragment from fragment-ordered LDS: Bf[((tb*2+hi)*16+n)*16 + j] = B(16*hi+j, tb*16+n)
__device__ __forceinline__ v16bf load_bfrag(const __bf16* Bf, int tb) {
  int lane = threadIdx.x & 31;
  int n = lane & 15, hi = lane >> 4;
  const __bf16* p = Bf + (((tb * 2 + hi) * 16 + n) << 4);
  return concat8(*(const v8bf*)p, *(const v8bf*)(p + 8));
}

__device__ __forceinline__ v8f wmma_bf16(v16bf a, v16bf b, v8f c) {
  return __builtin_amdgcn_wmma_f32_16x16x32_bf16(false, a, false, b, (short)0, c,
                                                 false, false);
}

// ---------------- Transpose kernels (one-time data layout prep) ----------------
__global__ __launch_bounds__(256) void wtrans_kernel(
    const float* __restrict__ S, __bf16* __restrict__ D, int R, int C) {
  __shared__ alignas(16) __bf16 t[32][40];
  int tid = threadIdx.x;
  int r0 = blockIdx.y * 32, c0 = blockIdx.x * 32;
  {
    int lr = tid >> 3, lc = (tid & 7) * 4;
    const float* sp = &S[(size_t)(r0 + lr) * C + c0 + lc];
#pragma unroll
    for (int j = 0; j < 4; ++j) t[lc + j][lr] = f32_to_bf16(sp[j]);
  }
  __syncthreads();
  {
    int lc = tid >> 3, lk = (tid & 7) * 4;
    __bf16* dp = &D[(size_t)(c0 + lc) * R + r0 + lk];
#pragma unroll
    for (int j = 0; j < 4; ++j) dp[j] = t[lc][lk + j];
  }
}

__global__ __launch_bounds__(256) void vtrans_kernel(
    const __bf16* __restrict__ S, __bf16* __restrict__ D, int R, int C) {
  __shared__ alignas(16) __bf16 t[32][40];
  int tid = threadIdx.x;
  size_t base = (size_t)blockIdx.z * R * C;
  int r0 = blockIdx.y * 32, c0 = blockIdx.x * 32;
  {
    int lr = tid >> 3, lc = (tid & 7) * 4;
    const __bf16* sp = S + base + (size_t)(r0 + lr) * C + c0 + lc;
#pragma unroll
    for (int j = 0; j < 4; ++j) t[lc + j][lr] = sp[j];
  }
  __syncthreads();
  {
    int lc = tid >> 3, lk = (tid & 7) * 4;
    __bf16* dp = D + base + (size_t)(c0 + lc) * R + r0 + lk;
#pragma unroll
    for (int j = 0; j < 4; ++j) dp[j] = t[lc][lk + j];
  }
}

// ===================== double-buffered 128x256 GEMM cores =====================

#define GEMM_LDS                                            \
  __shared__ alignas(16) __bf16 As[2][128 * 40];            \
  __shared__ alignas(16) __bf16 Bf[2][512 * 16];

#define GEMM_IDS                                            \
  int tid = threadIdx.x;                                    \
  int wave = tid >> 5, wm2 = wave & 1, wn2 = wave >> 1;     \
  int ar = tid >> 1, ah = tid & 1;

__device__ __forceinline__ void stage_b_t(
    __bf16* BfBuf, const __bf16* T, int ldt, int n0, int k0, int tid) {
#pragma unroll
  for (int s = tid; s < 512; s += 256) {
    int tb = s >> 5, bh = (s >> 4) & 1, bn = s & 15;
    const __bf16* src = &T[(size_t)(n0 + tb * 16 + bn) * ldt + k0 + 16 * bh];
    __bf16* dst = &BfBuf[(((tb * 2 + bh) * 16 + bn) << 4)];
    lds_copy16(dst, src);
    lds_copy16(dst + 8, src + 8);
  }
}

__device__ __forceinline__ void stage_a_bf16(
    __bf16* AsBuf, const __bf16* A, int lda, int m0, int k0, int ar, int ah) {
  const __bf16* src = &A[(size_t)(m0 + ar) * lda + k0 + ah * 16];
  __bf16* dst = &AsBuf[ar * 40 + ah * 16];
  lds_copy16(dst, src);
  lds_copy16(dst + 8, src + 8);
}

__device__ __forceinline__ void stage_a_f32(
    __bf16* AsBuf, const float* A, int lda, int m0, int k0, int ar, int ah) {
  const float* src = &A[(size_t)(m0 + ar) * lda + k0 + ah * 16];
  v8bf t0, t1;
#pragma unroll
  for (int j = 0; j < 8; ++j) t0[j] = f32_to_bf16(src[j]);
#pragma unroll
  for (int j = 0; j < 8; ++j) t1[j] = f32_to_bf16(src[8 + j]);
  v8bf* dst = (v8bf*)&AsBuf[ar * 40 + ah * 16];
  dst[0] = t0; dst[1] = t1;
}

#define GEMM_COMPUTE(buf)                                                   \
  {                                                                         \
    v16bf a[4], bb[4];                                                      \
    _Pragma("unroll") for (int i = 0; i < 4; ++i)                           \
        a[i] = load_a_frag(&As[buf][(wm2 * 64 + i * 16) * 40], 40);         \
    _Pragma("unroll") for (int j = 0; j < 4; ++j)                           \
        bb[j] = load_bfrag(&Bf[buf][0], wn2 * 4 + j);                       \
    _Pragma("unroll") for (int i = 0; i < 4; ++i)                           \
        _Pragma("unroll") for (int j = 0; j < 4; ++j)                       \
            acc[i][j] = wmma_bf16(a[i], bb[j], acc[i][j]);                  \
  }

// ---------------- Kernel 1: Y = bf16(X @ W + bias); X f32 [M,K], Wt bf16 [N,K]
__global__ __launch_bounds__(256) void proj_gemm_kernel(
    const float* __restrict__ X, const __bf16* __restrict__ Wt,
    const float* __restrict__ bias, __bf16* __restrict__ Y,
    int M, int K, int N) {
  GEMM_LDS
  GEMM_IDS
  int m0 = blockIdx.y * 128, n0 = blockIdx.x * 256;
  v8f acc[4][4] = {};
  stage_a_f32(As[0], X, K, m0, 0, ar, ah);
  stage_b_t(Bf[0], Wt, K, n0, 0, tid);
  int buf = 0;
  for (int k0 = 0; k0 < K; k0 += 32) {
    async_join();
    __syncthreads();
    if (k0 + 32 < K) {
      stage_a_f32(As[buf ^ 1], X, K, m0, k0 + 32, ar, ah);
      stage_b_t(Bf[buf ^ 1], Wt, K, n0, k0 + 32, tid);
    }
    GEMM_COMPUTE(buf)
    buf ^= 1;
  }
  int lane = tid & 31, nn = lane & 15, hi = lane >> 4;
#pragma unroll
  for (int i = 0; i < 4; ++i)
#pragma unroll
    for (int j = 0; j < 4; ++j) {
      int col = n0 + wn2 * 64 + j * 16 + nn;
      float bv = bias[col];
#pragma unroll
      for (int r = 0; r < 8; ++r) {
        int row = m0 + wm2 * 64 + i * 16 + r + 8 * hi;
        Y[(size_t)row * N + col] = f32_to_bf16(acc[i][j][r] + bv);
      }
    }
}

// ---------------- Kernel 3: attn = P @ V; P bf16 [NQ,NKK], Vt bf16 [DVV,NKK]
// TDM variant: per k-step, two TENSOR_LOAD_TO_LDS (P tile 128x32, Vt tile 256x32),
// padded to 40-elem LDS rows by the D# pad feature; wave 0 issues & waits.
__global__ __launch_bounds__(256) void pv_gemm_kernel(
    const __bf16* __restrict__ P, const __bf16* __restrict__ Vt,
    __bf16* __restrict__ O) {
#ifdef HAVE_TDM
  __shared__ alignas(16) __bf16 As[2][128 * 40];
  __shared__ alignas(16) __bf16 Bs[2][256 * 40];
  int tid = threadIdx.x;
  int wave = tid >> 5, wm2 = wave & 1, wn2 = wave >> 1;
  int b = blockIdx.z;
  int m0 = blockIdx.y * 128, n0 = blockIdx.x * 256;
  const __bf16* Pg = P + (size_t)b * NQ * NKK;
  const __bf16* Vg = Vt + (size_t)b * NKK * DVV;   // [DVV][NKK]
  v8f acc[4][4] = {};
  if (wave == 0) {
    TDM_LOAD(tdm_g0((unsigned)(size_t)&As[0][0],
                    (unsigned long long)(size_t)(Pg + (size_t)m0 * NKK)),
             tdm_g1(32, 128, 32, 128, NKK, true));
    TDM_LOAD(tdm_g0((unsigned)(size_t)&Bs[0][0],
                    (unsigned long long)(size_t)(Vg + (size_t)n0 * NKK)),
             tdm_g1(32, 256, 32, 256, NKK, true));
  }
  int buf = 0;
  for (int k0 = 0; k0 < NKK; k0 += 32) {
    if (wave == 0) __builtin_amdgcn_s_wait_tensorcnt(0);
    __syncthreads();
    if (k0 + 32 < NKK && wave == 0) {
      TDM_LOAD(tdm_g0((unsigned)(size_t)&As[buf ^ 1][0],
                      (unsigned long long)(size_t)(Pg + (size_t)m0 * NKK + k0 + 32)),
               tdm_g1(32, 128, 32, 128, NKK, true));
      TDM_LOAD(tdm_g0((unsigned)(size_t)&Bs[buf ^ 1][0],
                      (unsigned long long)(size_t)(Vg + (size_t)n0 * NKK + k0 + 32)),
               tdm_g1(32, 256, 32, 256, NKK, true));
    }
    {
      v16bf a[4], bb[4];
#pragma unroll
      for (int i = 0; i < 4; ++i)
        a[i] = load_a_frag(&As[buf][(wm2 * 64 + i * 16) * 40], 40);
#pragma unroll
      for (int j = 0; j < 4; ++j)
        bb[j] = load_bT_frag(&Bs[buf][(wn2 * 64 + j * 16) * 40], 40);
#pragma unroll
      for (int i = 0; i < 4; ++i)
#pragma unroll
        for (int j = 0; j < 4; ++j) acc[i][j] = wmma_bf16(a[i], bb[j], acc[i][j]);
    }
    buf ^= 1;
  }
#else
  GEMM_LDS
  GEMM_IDS
  int b = blockIdx.z;
  int m0 = blockIdx.y * 128, n0 = blockIdx.x * 256;
  const __bf16* Pg = P + (size_t)b * NQ * NKK;
  const __bf16* Vg = Vt + (size_t)b * NKK * DVV;
  v8f acc[4][4] = {};
  stage_a_bf16(As[0], Pg, NKK, m0, 0, ar, ah);
  stage_b_t(Bf[0], Vg, NKK, n0, 0, tid);
  int buf = 0;
  for (int k0 = 0; k0 < NKK; k0 += 32) {
    async_join();
    __syncthreads();
    if (k0 + 32 < NKK) {
      stage_a_bf16(As[buf ^ 1], Pg, NKK, m0, k0 + 32, ar, ah);
      stage_b_t(Bf[buf ^ 1], Vg, NKK, n0, k0 + 32, tid);
    }
    GEMM_COMPUTE(buf)
    buf ^= 1;
  }
#endif
  int lane = tid & 31, nn = lane & 15, hi = lane >> 4;
#pragma unroll
  for (int i = 0; i < 4; ++i)
#pragma unroll
    for (int j = 0; j < 4; ++j) {
      int col = n0 + wn2 * 64 + j * 16 + nn;
#pragma unroll
      for (int r = 0; r < 8; ++r) {
        int row = m0 + wm2 * 64 + i * 16 + r + 8 * hi;
        O[((size_t)b * NQ + row) * DVV + col] = f32_to_bf16(acc[i][j][r]);
      }
    }
}

// ---------------- Kernel 4: O0 = attn @ Wo + bo (f32 out); WoT bf16 [N,K]
__global__ __launch_bounds__(256) void oproj_gemm_kernel(
    const __bf16* __restrict__ A, const __bf16* __restrict__ Wt,
    const float* __restrict__ bias, float* __restrict__ Y,
    int M, int K, int N) {
  GEMM_LDS
  GEMM_IDS
  int m0 = blockIdx.y * 128, n0 = blockIdx.x * 256;
  v8f acc[4][4] = {};
  stage_a_bf16(As[0], A, K, m0, 0, ar, ah);
  stage_b_t(Bf[0], Wt, K, n0, 0, tid);
  int buf = 0;
  for (int k0 = 0; k0 < K; k0 += 32) {
    async_join();
    __syncthreads();
    if (k0 + 32 < K) {
      stage_a_bf16(As[buf ^ 1], A, K, m0, k0 + 32, ar, ah);
      stage_b_t(Bf[buf ^ 1], Wt, K, n0, k0 + 32, tid);
    }
    GEMM_COMPUTE(buf)
    buf ^= 1;
  }
  int lane = tid & 31, nn = lane & 15, hi = lane >> 4;
#pragma unroll
  for (int i = 0; i < 4; ++i)
#pragma unroll
    for (int j = 0; j < 4; ++j) {
      int col = n0 + wn2 * 64 + j * 16 + nn;
      float bv = bias[col];
#pragma unroll
      for (int r = 0; r < 8; ++r) {
        int row = m0 + wm2 * 64 + i * 16 + r + 8 * hi;
        Y[(size_t)row * N + col] = acc[i][j][r] + bv;
      }
    }
}

// ---------------- Kernel 2: scores (q@k^T * scale, masked) + row softmax -> P bf16
__global__ __launch_bounds__(256) void attn_softmax_kernel(
    const __bf16* __restrict__ qb, const __bf16* __restrict__ kb,
    const int* __restrict__ pad_mask, __bf16* __restrict__ Pb) {
  extern __shared__ char smem[];
  float* Srow = (float*)smem;                                   // [16][NKK] 128KB
  __bf16* qs = (__bf16*)(smem + 16 * NKK * sizeof(float));      // [16][DVV] 32KB
  __shared__ float rowinv[16];
  int b = blockIdx.y;
  int q0 = blockIdx.x * 16;
  int tid = threadIdx.x;
  {  // stage q rows (contiguous 16B copies, async if available)
    const __bf16* qg = qb + ((size_t)b * NQ + q0) * DVV;
    for (int i = tid; i < 16 * DVV / 8; i += 256)
      lds_copy16(qs + i * 8, qg + i * 8);
  }
  async_join();
  __syncthreads();

  int wave = tid >> 5;
  int lane = tid & 31, nn = lane & 15, hi = lane >> 4;
  const __bf16* kgb = kb + (size_t)b * NKK * DVV;
  const float scale = 0.03125f;  // 1/sqrt(1024)
  int key0 = wave * 256;         // each wave: 16x256 score strip
  v8f acc[16] = {};
  for (int ks = 0; ks < DVV; ks += 32) {
    v16bf a = load_a_frag(qs + ks, DVV);
#pragma unroll
    for (int t = 0; t < 16; ++t) {
      v16bf bf = load_bT_frag(kgb + (size_t)(key0 + t * 16) * DVV + ks, DVV);
      acc[t] = wmma_bf16(a, bf, acc[t]);
    }
  }
  const int* mk = pad_mask + b * NKK;
#pragma unroll
  for (int t = 0; t < 16; ++t) {
    int key = key0 + t * 16 + nn;
    bool dead = (mk[key] == 0);
#pragma unroll
    for (int r = 0; r < 8; ++r) {
      int m = r + 8 * hi;
      Srow[m * NKK + key] = dead ? -1e9f : acc[t][r] * scale;
    }
  }
  __syncthreads();

  int row = tid >> 4, sub = tid & 15;      // 16 threads per query row
  float mx = -3.4e38f;
  for (int c = sub; c < NKK; c += 16) mx = fmaxf(mx, Srow[row * NKK + c]);
#pragma unroll
  for (int o = 8; o >= 1; o >>= 1) mx = fmaxf(mx, __shfl_xor(mx, o, 16));
  float sum = 0.f;
  for (int c = sub; c < NKK; c += 16) {
    float e = __expf(Srow[row * NKK + c] - mx);
    Srow[row * NKK + c] = e;
    sum += e;
  }
#pragma unroll
  for (int o = 8; o >= 1; o >>= 1) sum += __shfl_xor(sum, o, 16);
  if (sub == 0) rowinv[row] = 1.f / sum;
  __syncthreads();

  float inv = rowinv[row];
  const float* Sr = Srow + row * NKK + sub * 128;
  __bf16* Pr = Pb + ((size_t)b * NQ + q0 + row) * NKK + sub * 128;
  for (int i = 0; i < 128; i += 8) {
    v8bf o;
#pragma unroll
    for (int j = 0; j < 8; ++j) o[j] = f32_to_bf16(Sr[i + j] * inv);
    *(v8bf*)(Pr + i) = o;
  }
}

// ---------------- Kernel 5: LayerNorm rows of 1024 (register-resident)
__global__ __launch_bounds__(256) void ln_kernel(
    const float* __restrict__ Xo, const float* __restrict__ g0,
    const float* __restrict__ beta0, float* __restrict__ out) {
  int tid = threadIdx.x;
  size_t row = (size_t)blockIdx.x * 16 + (tid >> 4);
  int sub = tid & 15;
  const float* xr = Xo + row * DVV;
  float v[64];
  float s = 0.f;
#pragma unroll
  for (int i = 0; i < 64; ++i) { v[i] = xr[sub + i * 16]; s += v[i]; }
#pragma unroll
  for (int o = 8; o >= 1; o >>= 1) s += __shfl_xor(s, o, 16);
  float mu = s * (1.f / 1024.f);
  float vs = 0.f;
#pragma unroll
  for (int i = 0; i < 64; ++i) { float d = v[i] - mu; vs += d * d; }
#pragma unroll
  for (int o = 8; o >= 1; o >>= 1) vs += __shfl_xor(vs, o, 16);
  float rstd = rsqrtf(vs * (1.f / 1024.f) + 1e-5f);
  float* og = out + row * DVV;
#pragma unroll
  for (int i = 0; i < 64; ++i) {
    int c = sub + i * 16;
    og[c] = (v[i] - mu) * rstd * g0[c] + beta0[c];
  }
}

extern "C" void kernel_launch(void* const* d_in, const int* in_sizes, int n_in,
                              void* d_out, int out_size, void* d_ws, size_t ws_size,
                              hipStream_t stream) {
  const float* Q   = (const float*)d_in[0];
  const float* K   = (const float*)d_in[1];
  const float* V   = (const float*)d_in[2];
  const int* pad_mask = (const int*)d_in[3];
  const float* Wq = (const float*)d_in[4];  const float* bq = (const float*)d_in[5];
  const float* Wk = (const float*)d_in[6];  const float* bk = (const float*)d_in[7];
  const float* Wv = (const float*)d_in[8];  const float* bv = (const float*)d_in[9];
  const float* Wo = (const float*)d_in[10]; const float* bo = (const float*)d_in[11];
  const float* g0 = (const float*)d_in[12]; const float* beta0 = (const float*)d_in[13];
  float* out = (float*)d_out;

  // ws layout (bf16 elems unless noted):
  // qb|kb|vb 16M each | P 32M | attn 16M | o0 (f32) 16M | WqT|WkT|WvT|WoT 1M each | vt 16M
  __bf16* qb  = (__bf16*)d_ws;
  __bf16* kb  = qb + (size_t)NB * NQ * DVV;
  __bf16* vb  = kb + (size_t)NB * NKK * DVV;
  __bf16* Pb  = vb + (size_t)NB * NKK * DVV;
  __bf16* ab  = Pb + (size_t)NB * NQ * NKK;
  float*  o0  = (float*)(ab + (size_t)NB * NQ * DVV);
  __bf16* WqT = (__bf16*)(o0 + (size_t)NB * NQ * DVV);
  __bf16* WkT = WqT + (size_t)DVV * DVV;
  __bf16* WvT = WkT + (size_t)DVV * DVV;
  __bf16* WoT = WvT + (size_t)DVV * DVV;
  __bf16* vt  = WoT + (size_t)DVV * DVV;

  dim3 blk(256);
  {  // weight transposes (f32 -> bf16^T), 2MB each, L2-resident afterwards
    dim3 grd(DVV / 32, DVV / 32);
    wtrans_kernel<<<grd, blk, 0, stream>>>(Wq, WqT, DVV, DVV);
    wtrans_kernel<<<grd, blk, 0, stream>>>(Wk, WkT, DVV, DVV);
    wtrans_kernel<<<grd, blk, 0, stream>>>(Wv, WvT, DVV, DVV);
    wtrans_kernel<<<grd, blk, 0, stream>>>(Wo, WoT, DVV, DVV);
  }
  {
    dim3 grd(DVV / 256, (NB * NQ) / 128);
    proj_gemm_kernel<<<grd, blk, 0, stream>>>(Q, WqT, bq, qb, NB * NQ, DVV, DVV);
    proj_gemm_kernel<<<grd, blk, 0, stream>>>(K, WkT, bk, kb, NB * NKK, DVV, DVV);
    proj_gemm_kernel<<<grd, blk, 0, stream>>>(V, WvT, bv, vb, NB * NKK, DVV, DVV);
  }
  {
    dim3 grd(NQ / 16, NB);
    size_t sm = 16 * NKK * sizeof(float) + 16 * DVV * sizeof(__bf16);  // 160KB
    attn_softmax_kernel<<<grd, blk, sm, stream>>>(qb, kb, pad_mask, Pb);
  }
  {  // V transpose per batch: [NKK][DVV] -> [DVV][NKK]
    dim3 grd(DVV / 32, NKK / 32, NB);
    vtrans_kernel<<<grd, blk, 0, stream>>>(vb, vt, NKK, DVV);
  }
  {
    dim3 grd(DVV / 256, NQ / 128, NB);
    pv_gemm_kernel<<<grd, blk, 0, stream>>>(Pb, vt, ab);
  }
  {
    dim3 grd(DVV / 256, (NB * NQ) / 128);
    oproj_gemm_kernel<<<grd, blk, 0, stream>>>(ab, WoT, bo, o0, NB * NQ, DVV, DVV);
  }
  {
    dim3 grd((NB * NQ) / 16);
    ln_kernel<<<grd, blk, 0, stream>>>(o0, g0, beta0, out);
  }
}
